// MultiHeadAttention_80307298501038
// MI455X (gfx1250) — matrix operand order repaired and checked by reading, not compiled, not run on hardware
//
#include <hip/hip_runtime.h>

// ---------------------------------------------------------------------------
// MultiHeadAttention for MI455X (gfx1250): all matmuls via v_wmma_f32_16x16x32_bf16
// B-tile staging via async global->LDS copies (ASYNCcnt) when available.
// ---------------------------------------------------------------------------
#define D_MODEL   1024
#define NUM_HEADS 16
#define DK        64
#define S_LEN     2048
#define BATCH     4
#define M_TOT     (BATCH * S_LEN)       // 8192
#define NEG_BIG   (-1000000000.0f)
#define KCHUNKS   (D_MODEL / 32)        // 32

typedef __bf16 bf16;
typedef __attribute__((ext_vector_type(16))) __bf16 bf16x16;
typedef __attribute__((ext_vector_type(8)))  __bf16 bf16x8;
typedef __attribute__((ext_vector_type(4)))  __bf16 bf16x4;
typedef __attribute__((ext_vector_type(8)))  float  f32x8;
typedef __attribute__((ext_vector_type(4)))  float  f32x4;
typedef __attribute__((ext_vector_type(4)))  int    i32x4;

#if defined(__AMDGCN__) && __has_builtin(__builtin_amdgcn_global_load_async_to_lds_b128)
#define HAVE_ASYNC_LDS 1
#else
#define HAVE_ASYNC_LDS 0
#endif

union BF16FRAG { bf16x16 v; bf16x8 h[2]; };

__device__ __forceinline__ f32x8 zero8() {
    f32x8 z = {0.f, 0.f, 0.f, 0.f, 0.f, 0.f, 0.f, 0.f};
    return z;
}

__device__ __forceinline__ f32x8 wmma_bf16(bf16x16 a, bf16x16 b, f32x8 c) {
    // 8 args: (neg_a, A, neg_b, B, c_mod, C, reuse_a, reuse_b)
    return __builtin_amdgcn_wmma_f32_16x16x32_bf16(false, a, false, b,
                                                   (short)0, c, false, false);
}

// 16B global -> LDS copy; async engine if the toolchain exposes it.
__device__ __forceinline__ void copy_b128_to_lds(const bf16* gsrc, bf16* ldst) {
#if HAVE_ASYNC_LDS
    __builtin_amdgcn_global_load_async_to_lds_b128(
        (__attribute__((address_space(1))) i32x4*)gsrc,
        (__attribute__((address_space(3))) i32x4*)ldst, 0, 0);
#else
    *(bf16x8*)ldst = *(const bf16x8*)gsrc;
#endif
}

__device__ __forceinline__ void wait_async() {
#if HAVE_ASYNC_LDS
#if __has_builtin(__builtin_amdgcn_s_wait_asynccnt)
    __builtin_amdgcn_s_wait_asynccnt(0);
#else
    asm volatile("s_wait_asynccnt 0" ::: "memory");
#endif
#endif
}

// A fragment (16x32 bf16, M x K) from a row-major row pointer (32 K-elems).
// lane m = lane&15 picks the row (caller), h = lane>>4.
// element e<8 -> K = 8h+e ; e>=8 -> K = 16+8h+(e-8). Two contiguous 16B loads.
__device__ __forceinline__ bf16x16 load_afrag(const bf16* rowptr, int h) {
    BF16FRAG f;
    f.h[0] = *(const bf16x8*)(rowptr + 8 * h);
    f.h[1] = *(const bf16x8*)(rowptr + 16 + 8 * h);
    return f.v;
}

// B fragment (32x16 bf16, K x N) from B^T stored row-major (row = N, 32 K-elems).
// lane n = lane&15 picks the row (caller), h = lane>>4; element e<8 -> K = 8h+e ; e>=8 -> K = 16+8h+(e-8).
__device__ __forceinline__ bf16x16 load_bfrag(const bf16* rowptr, int h) {
    BF16FRAG f;
    f.h[0] = *(const bf16x8*)(rowptr + 8 * h);
    f.h[1] = *(const bf16x8*)(rowptr + 16 + 8 * h);
    return f.v;
}

// ---------------------------------------------------------------------------
// One-shot f32 -> bf16 conversion (for the four weight matrices).
// ---------------------------------------------------------------------------
__global__ __launch_bounds__(256) void cvt_kernel(const float* __restrict__ src,
                                                  bf16* __restrict__ dst)
{
    const int i = (blockIdx.x * 256 + threadIdx.x) * 4;
    f32x4 v = *(const f32x4*)(src + i);
    bf16x4 b4;
    b4[0] = (bf16)v[0]; b4[1] = (bf16)v[1];
    b4[2] = (bf16)v[2]; b4[3] = (bf16)v[3];
    *(bf16x4*)(dst + i) = b4;
}

// ---------------------------------------------------------------------------
// Staging of one 32-wide K chunk into LDS (bf16, +8 pad per row).
//   A tile: 128 rows of X  (f32 -> bf16 for MODE<3, raw async bf16 for MODE 3)
//   B tile: 128 rows of Wb (bf16, raw async copy); Wb rows ARE B^T rows.
// ---------------------------------------------------------------------------
template<int MODE>
__device__ __forceinline__ void stage_tiles(
    const float* __restrict__ X, const bf16* __restrict__ Xb,
    const bf16* __restrict__ Wb,
    int m0, int n0, int k0, int tid,
    bf16 (*__restrict__ As)[40], bf16 (*__restrict__ Bs)[40])
{
    if (MODE < 3) {
        const int col4 = (tid & 7) * 4;
        const int rb   = tid >> 3;            // 0..31
#pragma unroll
        for (int rr = 0; rr < 128; rr += 32) {
            const int r = rb + rr;
            const float* src = X + (size_t)(m0 + r) * D_MODEL + k0 + col4;
            __builtin_prefetch(src + 64, 0, 1);   // chunk after next
            f32x4 v = *(const f32x4*)src;
            bf16x4 b4;
            b4[0] = (bf16)v[0]; b4[1] = (bf16)v[1];
            b4[2] = (bf16)v[2]; b4[3] = (bf16)v[3];
            *(bf16x4*)(&As[r][col4]) = b4;
        }
    } else {
        const int col8 = (tid & 3) * 8;       // 4 x 16B per 64B row
        const int rb   = tid >> 2;            // 0..63
#pragma unroll
        for (int rr = 0; rr < 128; rr += 64) {
            const int r = rb + rr;
            copy_b128_to_lds(Xb + (size_t)(m0 + r) * D_MODEL + k0 + col8,
                             &As[r][col8]);
        }
    }
    {
        const int col8 = (tid & 3) * 8;
        const int rb   = tid >> 2;            // 0..63
#pragma unroll
        for (int rr = 0; rr < 128; rr += 64) {
            const int r = rb + rr;
            copy_b128_to_lds(Wb + (size_t)(n0 + r) * D_MODEL + k0 + col8,
                             &Bs[r][col8]);
        }
    }
}

// ---------------------------------------------------------------------------
// GEMM: out = X @ W^T + bias.  X:[8192,1024], Wb:[1024,1024] bf16 row-major.
// MODE 0: out -> bf16 [B,H,S,64] (Q or K)   MODE 2: out -> bf16 [B,H,64,S] (V^T)
// MODE 3: A is bf16 (Xb), out -> f32 row-major [8192,1024]
// Block: 256 thr (8 waves), tile M=128 N=128; wave computes 32x64 (2x4 WMMA
// tiles). K staged 32 wide, double-buffered LDS, one barrier per chunk.
// ---------------------------------------------------------------------------
template<int MODE>
__global__ __launch_bounds__(256) void gemm_kernel(
    const float* __restrict__ X, const bf16* __restrict__ Xb,
    const bf16* __restrict__ Wb, const float* __restrict__ bias,
    bf16* __restrict__ outb, float* __restrict__ outf)
{
    __shared__ bf16 As[2][128][40];   // 2 x 10240 B
    __shared__ bf16 Bs[2][128][40];

    const int tid  = threadIdx.x;
    const int lane = tid & 31;
    const int wave = tid >> 5;
    const int wm   = wave & 3;        // 0..3 -> M offset 32*wm
    const int wn   = wave >> 2;       // 0..1 -> N offset 64*wn
    const int ln15 = lane & 15;
    const int h    = lane >> 4;
    const int m0   = blockIdx.x * 128;
    const int n0   = blockIdx.y * 128;

    f32x8 acc[2][4];
#pragma unroll
    for (int mt = 0; mt < 2; ++mt)
#pragma unroll
        for (int nt = 0; nt < 4; ++nt) acc[mt][nt] = zero8();

    stage_tiles<MODE>(X, Xb, Wb, m0, n0, 0, tid, As[0], Bs[0]);
    wait_async();
    __syncthreads();

    for (int kc = 0; kc < KCHUNKS; ++kc) {
        const int buf = kc & 1;
        if (kc + 1 < KCHUNKS)
            stage_tiles<MODE>(X, Xb, Wb, m0, n0, (kc + 1) * 32, tid,
                              As[buf ^ 1], Bs[buf ^ 1]);

        bf16x16 a[2], b[4];
#pragma unroll
        for (int mt = 0; mt < 2; ++mt)
            a[mt] = load_afrag(&As[buf][wm * 32 + mt * 16 + ln15][0], h);
#pragma unroll
        for (int nt = 0; nt < 4; ++nt)
            b[nt] = load_bfrag(&Bs[buf][wn * 64 + nt * 16 + ln15][0], h);
#pragma unroll
        for (int mt = 0; mt < 2; ++mt)
#pragma unroll
            for (int nt = 0; nt < 4; ++nt)
                acc[mt][nt] = wmma_bf16(a[mt], b[nt], acc[mt][nt]);

        wait_async();
        __syncthreads();
    }

    // ---- epilogue: bias + layout permute store ----
#pragma unroll
    for (int mt = 0; mt < 2; ++mt) {
#pragma unroll
        for (int nt = 0; nt < 4; ++nt) {
            const int gn = n0 + wn * 64 + nt * 16 + ln15;
            const float bv = bias[gn];
#pragma unroll
            for (int e = 0; e < 8; ++e) {
                const int gm    = m0 + wm * 32 + mt * 16 + e + 8 * h;
                const float val = acc[mt][nt][e] + bv;
                if (MODE == 3) {
                    outf[(size_t)gm * D_MODEL + gn] = val;
                } else {
                    const int b_  = gm >> 11;     // / 2048
                    const int s   = gm & 2047;
                    const int hh  = gn >> 6;      // / 64
                    const int dc  = gn & 63;
                    size_t off;
                    if (MODE == 2)
                        off = (((size_t)(b_ * NUM_HEADS + hh)) * DK + dc) * S_LEN + s;
                    else
                        off = (((size_t)(b_ * NUM_HEADS + hh)) * S_LEN + s) * DK + dc;
                    outb[off] = (bf16)val;
                }
            }
        }
    }
}

// ---------------------------------------------------------------------------
// Causal flash attention. Q,K: bf16 [B*H, S, 64]; Vt: bf16 [B*H, 64, S].
// Block = 256 thr (8 waves); wave w owns q rows [q0, q0+16); kv chunks of 32.
// ---------------------------------------------------------------------------
__global__ __launch_bounds__(256) void attn_kernel(
    const bf16* __restrict__ Q, const bf16* __restrict__ K,
    const bf16* __restrict__ Vt, bf16* __restrict__ AO)
{
    __shared__ bf16 P[8][16][40];   // per-wave P-tile staging (A layout source)

    const int tid  = threadIdx.x;
    const int lane = tid & 31;
    const int wave = tid >> 5;
    const int ln15 = lane & 15;
    const int h    = lane >> 4;
    const int bh   = blockIdx.y;                      // 0..63
    const int q0   = blockIdx.x * 128 + wave * 16;

    const bf16* Qbh = Q  + (size_t)bh * S_LEN * DK;
    const bf16* Kbh = K  + (size_t)bh * S_LEN * DK;
    const bf16* Vbh = Vt + (size_t)bh * DK * S_LEN;

    // Q fragments for this wave's 16 rows (held for the whole kv loop)
    bf16x16 qf0 = load_afrag(Qbh + (size_t)(q0 + ln15) * DK + 0,  h);
    bf16x16 qf1 = load_afrag(Qbh + (size_t)(q0 + ln15) * DK + 32, h);

    f32x8 o0 = zero8(), o1 = zero8(), o2 = zero8(), o3 = zero8();
    float mrow[8], lrow[8];
#pragma unroll
    for (int e = 0; e < 8; ++e) { mrow[e] = -1e30f; lrow[e] = 0.f; }

    const float scale = 0.125f;                       // 1/sqrt(64)
    const int kv_end  = q0 + 16;                      // causal: max col = q0+15

    for (int kv = 0; kv < kv_end; kv += 32) {
        // ---- scores: two 16-col subtiles, K-dim = 64 (2 WMMAs each) ----
        float sc[2][8];
#pragma unroll
        for (int c = 0; c < 2; ++c) {
            const bf16* krow = Kbh + (size_t)(kv + c * 16 + ln15) * DK;
            f32x8 z = zero8();
            z = wmma_bf16(qf0, load_bfrag(krow + 0,  h), z);
            z = wmma_bf16(qf1, load_bfrag(krow + 32, h), z);
            const int kvc = kv + c * 16 + ln15;       // this lane's kv column
            const bool need_mask = (kv + 31 > q0);
#pragma unroll
            for (int e = 0; e < 8; ++e) {
                float v = z[e] * scale;
                if (need_mask && kvc > q0 + e + 8 * h) v = NEG_BIG;
                sc[c][e] = v;
            }
        }

        // ---- online softmax (row = e + 8h, spread over 16 lanes per half) ----
        float corr[8];
#pragma unroll
        for (int e = 0; e < 8; ++e) {
            float mx = fmaxf(sc[0][e], sc[1][e]);
#pragma unroll
            for (int off = 8; off >= 1; off >>= 1)
                mx = fmaxf(mx, __shfl_xor(mx, off, 32));
            const float mnew = fmaxf(mrow[e], mx);
            corr[e] = __expf(mrow[e] - mnew);
            const float p0 = __expf(sc[0][e] - mnew);
            const float p1 = __expf(sc[1][e] - mnew);
            float rs = p0 + p1;
#pragma unroll
            for (int off = 8; off >= 1; off >>= 1)
                rs += __shfl_xor(rs, off, 32);
            lrow[e] = lrow[e] * corr[e] + rs;
            mrow[e] = mnew;
            sc[0][e] = p0; sc[1][e] = p1;
        }
#pragma unroll
        for (int e = 0; e < 8; ++e) {
            o0[e] *= corr[e]; o1[e] *= corr[e];
            o2[e] *= corr[e]; o3[e] *= corr[e];
        }

        // ---- relayout P (C-layout -> A-layout) via per-wave LDS tile ----
#pragma unroll
        for (int e = 0; e < 8; ++e) {
            P[wave][e + 8 * h][ln15]      = (bf16)sc[0][e];
            P[wave][e + 8 * h][16 + ln15] = (bf16)sc[1][e];
        }
        asm volatile("s_wait_dscnt 0" ::: "memory");  // LDS in-order per wave
        bf16x16 pf = load_afrag(&P[wave][ln15][0], h);

        // ---- O += P @ V  (4 d-tiles of 16) ----
        o0 = wmma_bf16(pf, load_bfrag(Vbh + (size_t)(0  + ln15) * S_LEN + kv, h), o0);
        o1 = wmma_bf16(pf, load_bfrag(Vbh + (size_t)(16 + ln15) * S_LEN + kv, h), o1);
        o2 = wmma_bf16(pf, load_bfrag(Vbh + (size_t)(32 + ln15) * S_LEN + kv, h), o2);
        o3 = wmma_bf16(pf, load_bfrag(Vbh + (size_t)(48 + ln15) * S_LEN + kv, h), o3);
    }

    // ---- finalize: divide by row sums, store to [B, S, H*64] bf16 ----
    const int b  = bh >> 4;
    const int hh = bh & 15;
    f32x8 os[4] = {o0, o1, o2, o3};
#pragma unroll
    for (int t = 0; t < 4; ++t) {
#pragma unroll
        for (int e = 0; e < 8; ++e) {
            const float val = os[t][e] / lrow[e];
            const int srow  = q0 + e + 8 * h;
            const size_t off = ((size_t)b * S_LEN + srow) * D_MODEL
                             + hh * 64 + t * 16 + ln15;
            AO[off] = (bf16)val;
        }
    }
}

// ---------------------------------------------------------------------------
extern "C" void kernel_launch(void* const* d_in, const int* in_sizes, int n_in,
                              void* d_out, int out_size, void* d_ws, size_t ws_size,
                              hipStream_t stream)
{
    const float* query = (const float*)d_in[0];
    const float* key_  = (const float*)d_in[1];
    const float* value = (const float*)d_in[2];
    // d_in[3] = causal mask (implicit in attn_kernel)
    const float* Wq = (const float*)d_in[4];  const float* bq = (const float*)d_in[5];
    const float* Wk = (const float*)d_in[6];  const float* bk = (const float*)d_in[7];
    const float* Wv = (const float*)d_in[8];  const float* bv = (const float*)d_in[9];
    const float* Wo = (const float*)d_in[10]; const float* bo = (const float*)d_in[11];
    float* out = (float*)d_out;

    char* ws = (char*)d_ws;
    const size_t seg  = (size_t)BATCH * NUM_HEADS * S_LEN * DK * sizeof(bf16); // 16 MB
    const size_t wseg = (size_t)D_MODEL * D_MODEL * sizeof(bf16);              // 2 MB
    bf16* Qb  = (bf16*)(ws + 0 * seg);
    bf16* Kb  = (bf16*)(ws + 1 * seg);
    bf16* Vtb = (bf16*)(ws + 2 * seg);
    bf16* AO  = (bf16*)(ws + 3 * seg);
    bf16* Wqb = (bf16*)(ws + 4 * seg + 0 * wseg);
    bf16* Wkb = (bf16*)(ws + 4 * seg + 1 * wseg);
    bf16* Wvb = (bf16*)(ws + 4 * seg + 2 * wseg);
    bf16* Wob = (bf16*)(ws + 4 * seg + 3 * wseg);

    const dim3 blk(256);
    const dim3 cgrid(D_MODEL * D_MODEL / (256 * 4));      // 1024
    const dim3 ggrid(M_TOT / 128, D_MODEL / 128);         // 64 x 8

    cvt_kernel<<<cgrid, blk, 0, stream>>>(Wq, Wqb);
    cvt_kernel<<<cgrid, blk, 0, stream>>>(Wk, Wkb);
    cvt_kernel<<<cgrid, blk, 0, stream>>>(Wv, Wvb);
    cvt_kernel<<<cgrid, blk, 0, stream>>>(Wo, Wob);

    gemm_kernel<0><<<ggrid, blk, 0, stream>>>(query, nullptr, Wqb, bq, Qb,  nullptr);
    gemm_kernel<0><<<ggrid, blk, 0, stream>>>(key_,  nullptr, Wkb, bk, Kb,  nullptr);
    gemm_kernel<2><<<ggrid, blk, 0, stream>>>(value, nullptr, Wvb, bv, Vtb, nullptr);

    attn_kernel<<<dim3(S_LEN / 128, BATCH * NUM_HEADS), blk, 0, stream>>>(Qb, Kb, Vtb, AO);

    gemm_kernel<3><<<ggrid, blk, 0, stream>>>(nullptr, AO, Wob, bo, nullptr, out);
}
